// FullAttention_24197845745692
// MI455X (gfx1250) — matrix operand order repaired and checked by reading, not compiled, not run on hardware
//
#include <hip/hip_runtime.h>

// ---------------------------------------------------------------------------
// CDNA5 (gfx1250) wave32 WMMA implementation of bidirectional full attention.
// All heavy math runs through v_wmma_f32_16x16x32_bf16 (f32 accumulate).
// ---------------------------------------------------------------------------

typedef __attribute__((ext_vector_type(16))) __bf16       v16bf;
typedef __attribute__((ext_vector_type(8)))  float        v8f;
typedef __attribute__((ext_vector_type(4)))  unsigned int v4u;

static constexpr int kB   = 16;
static constexpr int kP   = 4096;
static constexpr int kQ   = 512;
static constexpr int kDin = 512;
static constexpr int kH   = 256;
static constexpr int kDR  = 256;

__device__ __forceinline__ v8f wmma_bf16(v16bf a, v16bf b, v8f c) {
  // 8 args: (neg_a, A, neg_b, B, c_mod, C, reuse_a, reuse_b)
  return __builtin_amdgcn_wmma_f32_16x16x32_bf16(false, a, false, b, (short)0, c,
                                                 false, false);
}

union V16U { v16bf v; v4u u[2]; };

// A-fragment (16x32 bf16, lane = M row, two 8-elt chunks at k+8h and k+16+8h)
__device__ __forceinline__ v16bf frag_a_bf16(const __bf16* p, int ld, int lane) {
  int m = lane & 15, h = lane >> 4;
  const __bf16* q0 = p + (size_t)m * ld + 8 * h;
  V16U r;
  r.u[0] = *(const v4u*)q0;
  r.u[1] = *(const v4u*)(q0 + 16);
  return r.v;
}

// A-fragment built from f32 source (global or LDS) with on-the-fly bf16 convert
__device__ __forceinline__ v16bf frag_a_f32(const float* p, int ld, int lane) {
  int m = lane & 15, h = lane >> 4;
  const float* q0 = p + (size_t)m * ld + 8 * h;
  const float* q1 = q0 + 16;
  v16bf a;
#pragma unroll
  for (int j = 0; j < 8; ++j) { a[j] = (__bf16)q0[j]; a[8 + j] = (__bf16)q1[j]; }
  return a;
}

// B-fragment (32x16 bf16) sourced from [N][K] row-major storage; p points at
// element (n0,k0). Lane = N column, two 8-elt chunks at k0+8h and k0+16+8h.
__device__ __forceinline__ v16bf frag_b_bf16(const __bf16* p, int ld, int lane) {
  int n = lane & 15, h = lane >> 4;
  const __bf16* q = p + (size_t)n * ld + 8 * h;
  V16U r;
  r.u[0] = *(const v4u*)q;
  r.u[1] = *(const v4u*)(q + 16);
  return r.v;
}

// ---------------------------------------------------------------------------
// X[b][r][c] (f32) -> XT[b][c][r] (bf16): makes B-fragments contiguous loads.
// ---------------------------------------------------------------------------
__global__ void k_transpose_bf16(const float* __restrict__ X, __bf16* __restrict__ XT,
                                 int rows, int cols) {
  size_t per  = (size_t)rows * cols;
  size_t base = (size_t)blockIdx.y * per;
  size_t i    = (size_t)blockIdx.x * blockDim.x + threadIdx.x;
  if (i >= per) return;
  int r = (int)(i / cols), c = (int)(i % cols);
  XT[base + (size_t)c * rows + r] = (__bf16)X[base + i];
}

// ---------------------------------------------------------------------------
// U[r][h] = relu(X[r][:] @ W[:,h]) * (scaleD ? scaleD[h] : 1), stored bf16.
// Block = 8 waves; 32 rows/block; wave w owns columns [32w, 32w+32).
// 32x32 per-wave tile: 4 WMMAs per k-step with 2x A/B fragment reuse.
// ---------------------------------------------------------------------------
__global__ void __launch_bounds__(256)
k_gemm_relu(const float* __restrict__ X, const __bf16* __restrict__ WT,
            const float* __restrict__ scaleD, __bf16* __restrict__ U) {
  int lane = threadIdx.x & 31, wave = threadIdx.x >> 5;
  int row0 = blockIdx.x * 32;
  int n0   = wave * 32;
  const float* Xr0 = X + (size_t)row0 * kDin;
  const float* Xr1 = X + (size_t)(row0 + 16) * kDin;
  v8f c00 = {}, c01 = {}, c10 = {}, c11 = {};
  for (int k0 = 0; k0 < kDin; k0 += 32) {
    v16bf a0 = frag_a_f32(Xr0 + k0, kDin, lane);
    v16bf a1 = frag_a_f32(Xr1 + k0, kDin, lane);
    v16bf b0 = frag_b_bf16(WT + (size_t)n0 * kDin + k0, kDin, lane);
    v16bf b1 = frag_b_bf16(WT + (size_t)(n0 + 16) * kDin + k0, kDin, lane);
    c00 = wmma_bf16(a0, b0, c00);
    c01 = wmma_bf16(a0, b1, c01);
    c10 = wmma_bf16(a1, b0, c10);
    c11 = wmma_bf16(a1, b1, c11);
  }
  int n = lane & 15, h = lane >> 4;
  float s0 = scaleD ? scaleD[n0 + n] : 1.0f;
  float s1 = scaleD ? scaleD[n0 + 16 + n] : 1.0f;
#pragma unroll
  for (int v = 0; v < 8; ++v) {
    int r0 = row0 + v + 8 * h;
    int r1 = r0 + 16;
    U[(size_t)r0 * kH + n0 + n]      = (__bf16)(fmaxf(c00[v], 0.0f) * s0);
    U[(size_t)r0 * kH + n0 + 16 + n] = (__bf16)(fmaxf(c01[v], 0.0f) * s1);
    U[(size_t)r1 * kH + n0 + n]      = (__bf16)(fmaxf(c10[v], 0.0f) * s0);
    U[(size_t)r1 * kH + n0 + 16 + n] = (__bf16)(fmaxf(c11[v], 0.0f) * s1);
  }
}

// ---------------------------------------------------------------------------
// Direction 1: out[b,p,:] = softmax_q(Up[p]·Uq[q]) @ rep.
// One block per 16 passage rows; full 16x512 score row lives in LDS (32 KB).
// ---------------------------------------------------------------------------
__global__ void __launch_bounds__(256)
k_attn_p2q(const __bf16* __restrict__ Up, const __bf16* __restrict__ Uq,
           const __bf16* __restrict__ repT, const unsigned char* __restrict__ q_mask,
           float* __restrict__ out) {
  __shared__ float S[16][kQ];
  __shared__ float red[16][16];
  __shared__ float rowstat[16];
  int lane = threadIdx.x & 31, wave = threadIdx.x >> 5;
  int n = lane & 15, h = lane >> 4;
  int gr = blockIdx.x;          // row-tile index over B*P
  int b  = gr / (kP / 16);
  const __bf16* A  = Up + (size_t)gr * 16 * kH;
  const __bf16* Bq = Uq + (size_t)b * kQ * kH;
  const __bf16* R  = repT + (size_t)b * kDR * kQ;

  // warm L2->WGP path for phase-3 operand while scores are computed
  __builtin_prefetch(R + (size_t)(wave * 32) * kQ, 0, 1);

  // scores: wave owns q-columns [64w, 64w+64)
  v8f c[4] = {};
  for (int k0 = 0; k0 < kH; k0 += 32) {
    v16bf a = frag_a_bf16(A + k0, kH, lane);
#pragma unroll
    for (int t = 0; t < 4; ++t) {
      v16bf bb = frag_b_bf16(Bq + (size_t)(wave * 64 + t * 16) * kH + k0, kH, lane);
      c[t] = wmma_bf16(a, bb, c[t]);
    }
  }
#pragma unroll
  for (int t = 0; t < 4; ++t) {
    int q = wave * 64 + t * 16 + n;
    float msk = q_mask[(size_t)b * kQ + q] ? -1.0e30f : 0.0f;
#pragma unroll
    for (int v = 0; v < 8; ++v) S[v + 8 * h][q] = c[t][v] + msk;
  }
  __syncthreads();

  // row softmax over Q=512 (16 threads per row)
  int row = threadIdx.x >> 4, sub = threadIdx.x & 15;
  float pm = -3.0e38f;
  for (int j = sub; j < kQ; j += 16) pm = fmaxf(pm, S[row][j]);
  red[row][sub] = pm;
  __syncthreads();
  if (sub == 0) {
    float mm = -3.0e38f;
    for (int j = 0; j < 16; ++j) mm = fmaxf(mm, red[row][j]);
    rowstat[row] = mm;
  }
  __syncthreads();
  float mm = rowstat[row];
  float ps = 0.0f;
  for (int j = sub; j < kQ; j += 16) {
    float e = __expf(S[row][j] - mm);
    S[row][j] = e;
    ps += e;
  }
  red[row][sub] = ps;
  __syncthreads();
  if (sub == 0) {
    float ss = 0.0f;
    for (int j = 0; j < 16; ++j) ss += red[row][j];
    rowstat[row] = ss;
  }
  __syncthreads();

  // O = P @ rep : contraction over Q; wave owns d-columns [32w, 32w+32)
  v8f o0 = {}, o1 = {};
  for (int k0 = 0; k0 < kQ; k0 += 32) {
    v16bf a  = frag_a_f32(&S[0][0] + k0, kQ, lane);
    v16bf b0 = frag_b_bf16(R + (size_t)(wave * 32) * kQ + k0, kQ, lane);
    v16bf b1 = frag_b_bf16(R + (size_t)(wave * 32 + 16) * kQ + k0, kQ, lane);
    o0 = wmma_bf16(a, b0, o0);
    o1 = wmma_bf16(a, b1, o1);
  }
  float* O = out + (size_t)gr * 16 * kDR;
#pragma unroll
  for (int v = 0; v < 8; ++v) {
    int r = v + 8 * h;
    float inv = 1.0f / rowstat[r];
    O[(size_t)r * kDR + wave * 32 + n]      = o0[v] * inv;
    O[(size_t)r * kDR + wave * 32 + 16 + n] = o1[v] * inv;
  }
}

// ---------------------------------------------------------------------------
// Direction 2: out_q[b,q,:] = softmax_p(Uq[q]·Up[p]) @ rep_p.
// Flash-style streaming over P (chunks of 256) with online max/sum rescale.
// Next chunk is prefetched (global_prefetch_b8) while current one computes.
// ---------------------------------------------------------------------------
__global__ void __launch_bounds__(256)
k_attn_q2p(const __bf16* __restrict__ Uq, const __bf16* __restrict__ Up,
           const __bf16* __restrict__ rep_pT, const unsigned char* __restrict__ p_mask,
           float* __restrict__ outq) {
  __shared__ float S[16][256];
  __shared__ float red[16][16];
  __shared__ float mrun[16], lrun[16], rscale[16], newm[16];
  int lane = threadIdx.x & 31, wave = threadIdx.x >> 5;
  int n = lane & 15, h = lane >> 4;
  int row = threadIdx.x >> 4, sub = threadIdx.x & 15;
  int gr = blockIdx.x;          // row-tile index over B*Q
  int b  = gr / (kQ / 16);
  const __bf16* A   = Uq + (size_t)gr * 16 * kH;
  const __bf16* UpB = Up + (size_t)b * kP * kH;
  const __bf16* Rp  = rep_pT + (size_t)b * kDR * kP;

  v16bf afr[8];
#pragma unroll
  for (int kk = 0; kk < 8; ++kk) afr[kk] = frag_a_bf16(A + kk * 32, kH, lane);

  v8f o0 = {}, o1 = {};
  if (threadIdx.x < 16) { mrun[threadIdx.x] = -3.0e38f; lrun[threadIdx.x] = 0.0f; }
  __syncthreads();

  for (int p0 = 0; p0 < kP; p0 += 256) {
    // prefetch next chunk's operands (each lane pulls a distinct cacheline)
    if (p0 + 256 < kP) {
      __builtin_prefetch(UpB + (size_t)(p0 + 256 + wave * 32 + n) * kH + 16 * h, 0, 1);
      __builtin_prefetch(Rp + (size_t)(wave * 32 + n) * kP + p0 + 256 + 16 * h, 0, 1);
    }

    // S chunk [16][256]: wave owns p-columns [32w, 32w+32)
    v8f c0 = {}, c1 = {};
#pragma unroll
    for (int kk = 0; kk < 8; ++kk) {
      v16bf b0 = frag_b_bf16(UpB + (size_t)(p0 + wave * 32) * kH + kk * 32, kH, lane);
      v16bf b1 = frag_b_bf16(UpB + (size_t)(p0 + wave * 32 + 16) * kH + kk * 32, kH, lane);
      c0 = wmma_bf16(afr[kk], b0, c0);
      c1 = wmma_bf16(afr[kk], b1, c1);
    }
    {
      int pi0 = p0 + wave * 32 + n;
      float m0 = p_mask[(size_t)b * kP + pi0] ? -1.0e30f : 0.0f;
      float m1 = p_mask[(size_t)b * kP + pi0 + 16] ? -1.0e30f : 0.0f;
#pragma unroll
      for (int v = 0; v < 8; ++v) {
        S[v + 8 * h][wave * 32 + n]      = c0[v] + m0;
        S[v + 8 * h][wave * 32 + 16 + n] = c1[v] + m1;
      }
    }
    __syncthreads();

    // online softmax statistics
    float pm = -3.0e38f;
    for (int j = sub; j < 256; j += 16) pm = fmaxf(pm, S[row][j]);
    red[row][sub] = pm;
    __syncthreads();
    if (sub == 0) {
      float mm = mrun[row];
      for (int j = 0; j < 16; ++j) mm = fmaxf(mm, red[row][j]);
      newm[row]   = mm;
      rscale[row] = __expf(mrun[row] - mm);
      mrun[row]   = mm;
    }
    __syncthreads();
    float mm = newm[row];
    float ps = 0.0f;
    for (int j = sub; j < 256; j += 16) {
      float e = __expf(S[row][j] - mm);
      S[row][j] = e;
      ps += e;
    }
    red[row][sub] = ps;
    __syncthreads();
    if (sub == 0) {
      float ss = 0.0f;
      for (int j = 0; j < 16; ++j) ss += red[row][j];
      lrun[row] = lrun[row] * rscale[row] + ss;
    }
    __syncthreads();

    // rescale accumulator, then O += exp(S) @ rep_p (contraction over chunk)
#pragma unroll
    for (int v = 0; v < 8; ++v) {
      float sc = rscale[v + 8 * h];
      o0[v] *= sc;
      o1[v] *= sc;
    }
#pragma unroll
    for (int kk = 0; kk < 8; ++kk) {
      v16bf a  = frag_a_f32(&S[0][0] + kk * 32, 256, lane);
      v16bf b0 = frag_b_bf16(Rp + (size_t)(wave * 32) * kP + p0 + kk * 32, kP, lane);
      v16bf b1 = frag_b_bf16(Rp + (size_t)(wave * 32 + 16) * kP + p0 + kk * 32, kP, lane);
      o0 = wmma_bf16(a, b0, o0);
      o1 = wmma_bf16(a, b1, o1);
    }
    __syncthreads();   // protect S before next chunk overwrites it
  }

  float* O = outq + (size_t)gr * 16 * kDR;
#pragma unroll
  for (int v = 0; v < 8; ++v) {
    int r = v + 8 * h;
    float inv = 1.0f / lrun[r];
    O[(size_t)r * kDR + wave * 32 + n]      = o0[v] * inv;
    O[(size_t)r * kDR + wave * 32 + 16 + n] = o1[v] * inv;
  }
}

// ---------------------------------------------------------------------------
extern "C" void kernel_launch(void* const* d_in, const int* in_sizes, int n_in,
                              void* d_out, int out_size, void* d_ws, size_t ws_size,
                              hipStream_t stream) {
  (void)in_sizes; (void)n_in; (void)out_size; (void)ws_size;

  const float*         passage  = (const float*)d_in[0];
  const unsigned char* p_mask   = (const unsigned char*)d_in[1];
  const float*         question = (const float*)d_in[2];
  const unsigned char* q_mask   = (const unsigned char*)d_in[3];
  const float*         rep      = (const float*)d_in[4];
  const float*         rep_p    = (const float*)d_in[5];
  const float*         W        = (const float*)d_in[6];
  const float*         Dvec     = (const float*)d_in[7];

  // workspace layout (bf16 intermediates, ~72.3 MB total)
  char* ws = (char*)d_ws;
  __bf16* WT     = (__bf16*)ws; ws += (size_t)kH * kDin * 2;        // [H][Din]
  __bf16* Upb    = (__bf16*)ws; ws += (size_t)kB * kP * kH * 2;     // [B*P][H]
  __bf16* Uqb    = (__bf16*)ws; ws += (size_t)kB * kQ * kH * 2;     // [B*Q][H]
  __bf16* repT   = (__bf16*)ws; ws += (size_t)kB * kDR * kQ * 2;    // [B][DR][Q]
  __bf16* rep_pT = (__bf16*)ws; ws += (size_t)kB * kDR * kP * 2;    // [B][DR][P]

  float* out  = (float*)d_out;                     // [B,P,DR]
  float* outq = out + (size_t)kB * kP * kDR;       // [B,Q,DR]

  // 1. layout transforms (f32 -> transposed bf16)
  {
    dim3 g((kDin * kH + 255) / 256, 1);
    k_transpose_bf16<<<g, 256, 0, stream>>>(W, WT, kDin, kH);
  }
  {
    dim3 g((kQ * kDR + 255) / 256, kB);
    k_transpose_bf16<<<g, 256, 0, stream>>>(rep, repT, kQ, kDR);
  }
  {
    dim3 g((kP * kDR + 255) / 256, kB);
    k_transpose_bf16<<<g, 256, 0, stream>>>(rep_p, rep_pT, kP, kDR);
  }

  // 2. projections: Up = relu(passage@W), Uq = relu(question@W)*D  (bf16 out)
  k_gemm_relu<<<dim3(kB * kP / 32), 256, 0, stream>>>(passage, WT, nullptr, Upb);
  k_gemm_relu<<<dim3(kB * kQ / 32), 256, 0, stream>>>(question, WT, Dvec, Uqb);

  // 3. attention, both directions
  k_attn_p2q<<<dim3(kB * kP / 16), 256, 0, stream>>>(Upb, Uqb, repT, q_mask, out);
  k_attn_q2p<<<dim3(kB * kQ / 16), 256, 0, stream>>>(Uqb, Upb, rep_pT, p_mask, outq);
}
